// EnhancedFFTPeriodDetection_68521908240461
// MI455X (gfx1250) — matrix operand order repaired and checked by reading, not compiled, not run on hardware
//
#include <hip/hip_runtime.h>
#include <hip/hip_bf16.h>

typedef __attribute__((ext_vector_type(16))) _Float16 v16h;
typedef __attribute__((ext_vector_type(8)))  float    v8f;

#define T_LEN 4096
#define C_CH  64
#define BATCH 64
#define NFREQ 2049          // T/2 + 1
#define F_PAD 2064          // 129 * 16
#define NCAND 6             // 2 * TOP_K
#define HALF_T 2048
#define NB    4             // batches per DFT block (A-fragment reuse factor)

// ---------------------------------------------------------------------------
// Kernel 0: zero the amplitude accumulator, build the f16 cosine table.
// cos(2*pi*f*t/4096) = costab[(f*t) & 4095]; sin is a -1024 index shift.
// ---------------------------------------------------------------------------
__global__ void init_kernel(float* __restrict__ amps, _Float16* __restrict__ costab) {
    int i = blockIdx.x * blockDim.x + threadIdx.x;
    if (i < BATCH * F_PAD) amps[i] = 0.0f;
    if (i < T_LEN) {
        float th = 6.28318530717958647692f * (float)i / (float)T_LEN;
        costab[i] = (_Float16)cosf(th);
    }
}

// ---------------------------------------------------------------------------
// Kernel 0b: x[b][t][c] f32 -> x16[b][c][t] f16 (LDS-tiled transpose so both
// the global read and the global write are coalesced).
// ---------------------------------------------------------------------------
__global__ void __launch_bounds__(256)
convert_x_kernel(const float* __restrict__ x, _Float16* __restrict__ x16) {
    const int b   = blockIdx.y;
    const int t0  = blockIdx.x * 64;
    const int tid = threadIdx.x;
    __shared__ _Float16 tile[C_CH][64 + 2];   // +2 halves pad vs bank conflicts

    for (int i = tid; i < 64 * C_CH; i += 256) {
        int tt = i >> 6;                      // time within tile
        int c  = i & 63;                      // channel fastest -> coalesced read
        tile[c][tt] = (_Float16)x[((size_t)b * T_LEN + (t0 + tt)) * C_CH + c];
    }
    __syncthreads();
    for (int i = tid; i < 64 * C_CH; i += 256) {
        int c  = i >> 6;
        int tt = i & 63;                      // time fastest -> coalesced write
        x16[((size_t)(b * C_CH + c)) * T_LEN + t0 + tt] = tile[c][tt];
    }
}

// ---------------------------------------------------------------------------
// Kernel 1: DFT magnitudes via WMMA, A-fragment reuse across NB=4 batches.
// Grid: (129 freq tiles, 16 batch groups), block = 128 threads = 4 waves.
// Each wave owns [16 freq x 16 chan] Re/Im tiles for 4 batches. Per K-step:
// build the (batch-independent) cos/sin A fragments once, then 8 WMMAs.
// ---------------------------------------------------------------------------
template <bool F16PATH>
__global__ void __launch_bounds__(128)
dft_amps_wmma(const float* __restrict__ x,
              const _Float16* __restrict__ x16,
              const _Float16* __restrict__ costab,
              float* __restrict__ amps) {
    const int ftile = blockIdx.x;           // 0..128
    const int b0    = blockIdx.y * NB;      // batch group base
    const int wave  = threadIdx.x >> 5;     // 0..3  -> channel group
    const int lane  = threadIdx.x & 31;
    const int hi    = lane >> 4;            // half-wave select
    const int l15   = lane & 15;

    __shared__ _Float16 s_cos[T_LEN];       // 8 KB staged twiddle table
    for (int i = threadIdx.x; i < T_LEN; i += blockDim.x) s_cos[i] = costab[i];
    __syncthreads();
    const char* scb = (const char*)s_cos;   // byte-addressed LDS view

    const int f  = ftile * 16 + l15;        // A-matrix row (frequency)
    const int c0 = wave * 16;               // channel group base

    // Per-lane constant LDS *byte* offsets for this kernel's contiguous A k order (same as B):
    // element j: K = j (+16 for lanes 16-31)
    int fk2[16];
#pragma unroll
    for (int j = 0; j < 16; ++j) {
        int p = j >> 1, h = j & 1;
        int k = (p < 4) ? (2 * p + h) : (8 + 2 * (p - 4) + h);
        k += hi ? 16 : 0;
        fk2[j] = ((f * k) & (T_LEN - 1)) * 2;
    }
    const int step2 = ((f * 32) & (T_LEN - 1)) * 2;
    int base2 = 0;

    v8f acc_re[NB] = {};
    v8f acc_im[NB] = {};

    const _Float16* xcb[NB];
    const float*    xfb[NB];
#pragma unroll
    for (int g = 0; g < NB; ++g) {
        if (F16PATH)
            xcb[g] = x16 + (size_t)((b0 + g) * C_CH + c0 + l15) * T_LEN + (hi ? 16 : 0);
        else
            xfb[g] = x + (size_t)(b0 + g) * T_LEN * C_CH + (c0 + l15);
    }

    for (int t0 = 0; t0 < T_LEN; t0 += 32) {
        // Batch-independent twiddle fragments (built once, used by 8 WMMAs)
        v16h a_cos, a_sin;
#pragma unroll
        for (int j = 0; j < 16; ++j) {
            int ac = (base2 + fk2[j]) & (2 * T_LEN - 1);
            a_cos[j] = *(const _Float16*)(scb + ac);
            a_sin[j] = *(const _Float16*)(scb + ((ac + 6144) & (2 * T_LEN - 1)));
        }
        // B fragments: 32x16 (K x N); lanes 0-15 hold K=j, lanes 16-31 K=16+j
        v16h bf[NB];
#pragma unroll
        for (int g = 0; g < NB; ++g) {
            if (F16PATH) {
                bf[g] = *reinterpret_cast<const v16h*>(xcb[g] + t0);  // 32B aligned
            } else {
#pragma unroll
                for (int j = 0; j < 16; ++j) {
                    int k = j + (hi ? 16 : 0);
                    bf[g][j] = (_Float16)xfb[g][(size_t)(t0 + k) * C_CH];
                }
            }
        }
#pragma unroll
        for (int g = 0; g < NB; ++g) {
            acc_re[g] = __builtin_amdgcn_wmma_f32_16x16x32_f16(
                false, a_cos, false, bf[g], (short)0, acc_re[g], false, false);
            acc_im[g] = __builtin_amdgcn_wmma_f32_16x16x32_f16(
                false, a_sin, false, bf[g], (short)0, acc_im[g], false, false);
        }
        base2 = (base2 + step2) & (2 * T_LEN - 1);
    }

    // |X[f,c]| per element; C/D layout: VGPR r -> row M=r+8*hi, N = lane&15.
    // Sum over this wave's 16 channels = reduce across each 16-lane half.
#pragma unroll
    for (int g = 0; g < NB; ++g) {
#pragma unroll
        for (int r = 0; r < 8; ++r) {
            float m = sqrtf(acc_re[g][r] * acc_re[g][r] + acc_im[g][r] * acc_im[g][r]);
            m += __shfl_xor(m, 8, 16);
            m += __shfl_xor(m, 4, 16);
            m += __shfl_xor(m, 2, 16);
            m += __shfl_xor(m, 1, 16);
            if (l15 == 0) {
                int fr = ftile * 16 + r + hi * 8;
                if (fr < NFREQ) atomicAdd(&amps[(b0 + g) * F_PAD + fr], m);
            }
        }
    }
}

// ---------------------------------------------------------------------------
// Kernel 2: per-batch top-6 bins -> integer candidate periods + range flags.
// ---------------------------------------------------------------------------
__global__ void topk_kernel(const float* __restrict__ amps,
                            int* __restrict__ cand_p,
                            int* __restrict__ cand_flag) {
    const int b   = blockIdx.x;
    const int tid = threadIdx.x;            // 256 threads
    __shared__ float sval[256];
    __shared__ int   sidx[256];
    __shared__ int   picked[NCAND];

    for (int k = 0; k < NCAND; ++k) {
        float best = -1e30f; int bi = -1;
        for (int fq = tid; fq < NFREQ; fq += 256) {
            bool skip = false;
            for (int j = 0; j < k; ++j) skip |= (picked[j] == fq);
            float v = amps[b * F_PAD + fq];
            if (!skip && v > best) { best = v; bi = fq; }
        }
        sval[tid] = best; sidx[tid] = bi;
        __syncthreads();
        for (int s = 128; s; s >>= 1) {
            if (tid < s) {
                bool take = (sval[tid + s] > sval[tid]) ||
                            (sval[tid + s] == sval[tid] && sidx[tid + s] >= 0 &&
                             sidx[tid + s] < sidx[tid]);
                if (take) { sval[tid] = sval[tid + s]; sidx[tid] = sidx[tid + s]; }
            }
            __syncthreads();
        }
        if (tid == 0) picked[k] = sidx[0];
        __syncthreads();
    }
    if (tid < NCAND) {
        int idx = picked[tid];
        float period_f = (float)T_LEN / (float)(idx + 1);
        float pf = floorf(period_f);
        int in_range = (period_f >= 2.0f) && (period_f <= (float)HALF_T) &&
                       (pf >= 2.0f) && (pf <= (float)HALF_T);
        int p_int = (int)pf;
        p_int = p_int < 2 ? 2 : (p_int > HALF_T ? HALF_T : p_int);
        cand_p[b * NCAND + tid]    = p_int;
        cand_flag[b * NCAND + tid] = in_range;
    }
}

// ---------------------------------------------------------------------------
// Kernel 3: autocorrelation cosine-similarity score per (batch, candidate).
// One block per task; per-segment dot/norm bins in LDS.
// ---------------------------------------------------------------------------
__global__ void __launch_bounds__(256)
autocorr_kernel(const float* __restrict__ x,
                const int* __restrict__ cand_p,
                float* __restrict__ scores) {
    const int b    = blockIdx.x / NCAND;
    const int cand = blockIdx.x % NCAND;
    const int tid  = threadIdx.x;           // 256 threads

    __shared__ float sdot[HALF_T];
    __shared__ float sq[HALF_T];
    __shared__ float red[256];

    const int p    = cand_p[b * NCAND + cand];   // guaranteed in [2, 2048]
    const int nper = T_LEN / p;

    for (int i = tid; i < HALF_T; i += 256) { sdot[i] = 0.0f; sq[i] = 0.0f; }
    __syncthreads();

    const float* xb = x + (size_t)b * T_LEN * C_CH;
    for (int t = tid; t < T_LEN; t += 256) {
        const float* row = xb + (size_t)t * C_CH;
        bool hasNext = (t + p) < T_LEN;
        const float* row2 = xb + (size_t)(hasNext ? t + p : t) * C_CH;
        __builtin_prefetch(row2, 0, 1);     // global_prefetch_b8
        float q = 0.0f, r = 0.0f;
#pragma unroll 8
        for (int c = 0; c < C_CH; ++c) {
            float a = row[c];
            q += a * a;
            r += a * row2[c];
        }
        if (!hasNext) r = 0.0f;
        int seg = t / p;
        if (seg < HALF_T) { atomicAdd(&sdot[seg], r); atomicAdd(&sq[seg], q); }
    }
    __syncthreads();

    float local = 0.0f;
    for (int i = tid; i < nper - 1; i += 256) {
        float na = sqrtf(fmaxf(sq[i], 0.0f));
        float nb = sqrtf(fmaxf(sq[i + 1], 0.0f));
        local += sdot[i] / fmaxf(na * nb, 1e-8f);
    }
    red[tid] = local;
    __syncthreads();
    for (int s = 128; s; s >>= 1) {
        if (tid < s) red[tid] += red[tid + s];
        __syncthreads();
    }
    if (tid == 0) {
        float denom = (float)((nper - 1) > 1 ? (nper - 1) : 1);
        scores[b * NCAND + cand] = (nper >= 2) ? red[0] / denom : 0.0f;
    }
}

// ---------------------------------------------------------------------------
// Kernel 4: valid-first stable selection, fallback periods, write outputs.
// out[0..191] = periods[64,3]; out[192..383] = weights[64,3].
// ---------------------------------------------------------------------------
__global__ void select_kernel(const int* __restrict__ cand_p,
                              const int* __restrict__ cand_flag,
                              const float* __restrict__ scores,
                              float* __restrict__ out) {
    int b = blockIdx.x * blockDim.x + threadIdx.x;
    if (b >= BATCH) return;
    const int common[3] = { T_LEN / 4, T_LEN / 3, T_LEN / 2 };  // int(T/1.5)=2730 > 2048 filtered
    float pers[3], wts[3];
    int cnt = 0;
    for (int j = 0; j < NCAND && cnt < 3; ++j) {
        if (cand_flag[b * NCAND + j] && scores[b * NCAND + j] > 0.2f) {
            pers[cnt] = (float)cand_p[b * NCAND + j];
            wts[cnt]  = 1.0f;
            ++cnt;
        }
    }
    for (int j = cnt; j < 3; ++j) {
        int k = j - cnt; if (k > 2) k = 2;
        pers[j] = (float)common[k];
        wts[j]  = 0.5f;
    }
    for (int j = 0; j < 3; ++j) {
        out[b * 3 + j]             = pers[j];
        out[BATCH * 3 + b * 3 + j] = wts[j];
    }
}

// ---------------------------------------------------------------------------
extern "C" void kernel_launch(void* const* d_in, const int* in_sizes, int n_in,
                              void* d_out, int out_size, void* d_ws, size_t ws_size,
                              hipStream_t stream) {
    (void)in_sizes; (void)n_in; (void)out_size;
    const float* x  = (const float*)d_in[0];
    float* out      = (float*)d_out;
    char* ws        = (char*)d_ws;

    const size_t X16_BYTES  = (size_t)BATCH * C_CH * T_LEN * sizeof(_Float16); // 32 MB
    const size_t AMPS_BYTES = (size_t)BATCH * F_PAD * sizeof(float);           // 516 KB
    const size_t COS_BYTES  = (size_t)T_LEN * sizeof(_Float16);                // 8 KB
    const size_t SMALL      = (size_t)BATCH * NCAND * sizeof(int);             // 1.5 KB each
    const bool   bigws      = ws_size >= X16_BYTES + AMPS_BYTES + COS_BYTES + 3 * SMALL;

    size_t off = bigws ? X16_BYTES : 0;
    _Float16* x16       = (_Float16*)ws;                 // only used when bigws
    float*    amps      = (float*)(ws + off);            off += AMPS_BYTES;
    _Float16* costab    = (_Float16*)(ws + off);         off += COS_BYTES;
    int*      cand_p    = (int*)(ws + off);              off += SMALL;
    int*      cand_flag = (int*)(ws + off);              off += SMALL;
    float*    scores    = (float*)(ws + off);

    int init_n = BATCH * F_PAD;
    init_kernel<<<(init_n + 255) / 256, 256, 0, stream>>>(amps, costab);

    if (bigws) {
        convert_x_kernel<<<dim3(T_LEN / 64, BATCH), 256, 0, stream>>>(x, x16);
        dft_amps_wmma<true><<<dim3(F_PAD / 16, BATCH / NB), 128, 0, stream>>>(x, x16, costab, amps);
    } else {
        dft_amps_wmma<false><<<dim3(F_PAD / 16, BATCH / NB), 128, 0, stream>>>(x, nullptr, costab, amps);
    }

    topk_kernel<<<BATCH, 256, 0, stream>>>(amps, cand_p, cand_flag);

    autocorr_kernel<<<BATCH * NCAND, 256, 0, stream>>>(x, cand_p, scores);

    select_kernel<<<1, 64, 0, stream>>>(cand_p, cand_flag, scores, out);
}
